// Quantize_46282567582172
// MI455X (gfx1250) — matrix operand order repaired and checked by reading, not compiled, not run on hardware
//
#include <hip/hip_runtime.h>

// ---------------------------------------------------------------------------
// CDNA5 (gfx1250) fused quantize kernel, round 6.
//   logits = x@W^T + b  via V_WMMA_F32_16X16X32_BF16, hi/lo split (bf16x3)
//   out    = one-hot(argmax(logits+gumbel)) gather from vars
//   y2     = softmax(mean(logits)) via deterministic two-phase reduction
// Round-6 changes: explicit per-nt running B pointers (+64B/K-step) replace
// base+offset math -> removes the spilled 64-bit offset temps and the
// per-use v_add_nc_u64; x-tile register prefetch dropped (-16 VGPRs).
// ---------------------------------------------------------------------------

typedef __attribute__((ext_vector_type(16))) __bf16        v16bf;
typedef __attribute__((ext_vector_type(8)))  float         v8f;
typedef __attribute__((ext_vector_type(8)))  unsigned int  v8u;

union FragU {
    v8u   u;
    uint4 q[2];
    v16bf b;
};

#define BT_ROWS   32768   // 16*2048
#define HID       768
#define NCOL      640
#define ENTRYV    320
#define VARD      384
#define BM        128     // rows per block
#define MT        8       // M-tiles per wave
#define KSTEP     32
#define NSTEPS    (HID / KSTEP)    // 24
#define NT_WAVE   5       // 40 N-tiles / 8 waves
#define NBLOCKS   (BT_ROWS / BM)   // 256

// Split f32 pair (even K, odd K) into packed bf16 hi (truncate) + bf16 residual lo.
// Packing: even K -> bits[15:0], odd K -> bits[31:16] (ISA §7.12.2 pair order).
__device__ __forceinline__ void bsplit_pack(float e, float o, unsigned &hi, unsigned &lo) {
    unsigned eh = __float_as_uint(e) >> 16;
    unsigned oh = __float_as_uint(o) >> 16;
    float re = e - __uint_as_float(eh << 16);
    float ro = o - __uint_as_float(oh << 16);
    hi = (oh << 16) | eh;
    lo = ((__float_as_uint(ro) >> 16) << 16) | (__float_as_uint(re) >> 16);
}

// ---------------------------------------------------------------------------
// Kernel 1: split W[640][768] into packed hi/lo bf16 pair arrays [640][384] u32,
// laid out so a B-fragment is two uint4 loads per lane.
// ---------------------------------------------------------------------------
__global__ __launch_bounds__(256)
void wsplit_kernel(const float* __restrict__ W,
                   unsigned* __restrict__ Wh,
                   unsigned* __restrict__ Wl) {
    int p = blockIdx.x * 256 + threadIdx.x;          // pair index: n*384 + k/2
    if (p >= NCOL * (HID / 2)) return;
    float2 wv = *(const float2*)(W + 2 * p);
    unsigned h, l;
    bsplit_pack(wv.x, wv.y, h, l);
    Wh[p] = h;
    Wl[p] = l;
}

// ---------------------------------------------------------------------------
// Kernel 2: fused GEMM (bf16x3 WMMA) + register-resident epilogue.
// Block: 256 threads (8 waves), 128 rows x 640 cols.
//   wave w owns N columns [80w, 80w+80) for all 8 M-tiles -> acc[8][5] (v8f).
// ---------------------------------------------------------------------------
__global__ __launch_bounds__(256)
void qgemm_kernel(const float*    __restrict__ x,
                  const unsigned* __restrict__ Wh,
                  const unsigned* __restrict__ Wl,
                  const float*    __restrict__ bvec,
                  const float*    __restrict__ vars,
                  const float*    __restrict__ gumbel,
                  float*          __restrict__ out,
                  float*          __restrict__ partial) {
    __shared__ unsigned Ah[2][BM * 16];   // double-buffered bf16-hi pairs (2x8KB)
    __shared__ unsigned Al[2][BM * 16];   // double-buffered bf16-lo pairs (2x8KB)
    __shared__ float    cval[BM][8];      // per-row per-wave argmax candidate value
    __shared__ int      cidx[BM][8];      // per-row per-wave argmax candidate index

    const int tid  = threadIdx.x;
    const int wave = tid >> 5;
    const int lane = tid & 31;
    const int half = lane >> 4;           // ISA lane-half (K interleave select)
    const int mrow = lane & 15;           // M (A) / N (B,C,D) index in tile
    const int r0   = blockIdx.x * BM;

    v8f acc[MT][NT_WAVE];
    #pragma unroll
    for (int m = 0; m < MT; ++m)
        #pragma unroll
        for (int n = 0; n < NT_WAVE; ++n)
            #pragma unroll
            for (int v = 0; v < 8; ++v)
                acc[m][n][v] = 0.0f;

    // per-nt running B pointers (advanced +16 u32 = one K-step per iteration)
    const unsigned* pWh[NT_WAVE];
    const unsigned* pWl[NT_WAVE];
    #pragma unroll
    for (int nt = 0; nt < NT_WAVE; ++nt) {
        const unsigned off =
            (unsigned)((wave * 80 + nt * 16 + mrow) * (HID / 2) + 8 * half);
        pWh[nt] = Wh + off;
        pWl[nt] = Wl + off;
    }

    // cooperative A staging: 256 thr x 16 floats = 128x32 tile
    const int    srow = tid >> 1;                  // 0..127
    const int    sc   = (tid & 1) << 4;            // 0 or 16 (half row of K)
    const float* xrow = x + (size_t)(r0 + srow) * HID + sc;
    const int    sbase = srow * 16 + (sc >> 1);    // u32 pair base in LDS tile

    for (int ks = 0; ks < NSTEPS; ++ks) {
        const int k0  = ks * KSTEP;
        const int buf = ks & 1;

        // stage + split A tile into LDS (16 floats -> 8 hi + 8 lo pairs)
        {
            uint4 h0, l0, h1, l1;
            float4 a0 = *(const float4*)(xrow + k0);
            float4 a1 = *(const float4*)(xrow + k0 + 4);
            float4 a2 = *(const float4*)(xrow + k0 + 8);
            float4 a3 = *(const float4*)(xrow + k0 + 12);
            bsplit_pack(a0.x, a0.y, h0.x, l0.x);
            bsplit_pack(a0.z, a0.w, h0.y, l0.y);
            bsplit_pack(a1.x, a1.y, h0.z, l0.z);
            bsplit_pack(a1.z, a1.w, h0.w, l0.w);
            bsplit_pack(a2.x, a2.y, h1.x, l1.x);
            bsplit_pack(a2.z, a2.w, h1.y, l1.y);
            bsplit_pack(a3.x, a3.y, h1.z, l1.z);
            bsplit_pack(a3.z, a3.w, h1.w, l1.w);
            uint4* AhP = (uint4*)(Ah[buf] + sbase);
            uint4* AlP = (uint4*)(Al[buf] + sbase);
            AhP[0] = h0; AhP[1] = h1;
            AlP[0] = l0; AlP[1] = l1;
        }
        __syncthreads();

        // A fragments (contiguous K layout: pairs {8h+v}, v=0..7)
        FragU aH[MT], aL[MT];
        #pragma unroll
        for (int m = 0; m < MT; ++m) {
            const uint4* pAh = (const uint4*)(Ah[buf] + (m * 16 + mrow) * 16);
            const uint4* pAl = (const uint4*)(Al[buf] + (m * 16 + mrow) * 16);
            aH[m].q[0] = pAh[2 * half]; aH[m].q[1] = pAh[2 * half + 1];
            aL[m].q[0] = pAl[2 * half]; aL[m].q[1] = pAl[2 * half + 1];
        }

        // B fragments from running pointers.
        // bH: 2-slot register prefetch (slot nt+1 in flight during nt's WMMAs)
        // bL: single slot, loaded at top of iteration, consumed 16 WMMAs later
        FragU bH[2];
        {
            const uint4* p = (const uint4*)pWh[0];
            bH[0].q[0] = p[0]; bH[0].q[1] = p[1];
        }
        #pragma unroll
        for (int nt = 0; nt < NT_WAVE; ++nt) {
            const int cur = nt & 1;
            const int nxt = cur ^ 1;
            FragU bL;
            {
                const uint4* p = (const uint4*)pWl[nt];
                bL.q[0] = p[0]; bL.q[1] = p[1];
                pWl[nt] += 16;
            }
            if (nt + 1 < NT_WAVE) {   // prefetch bH for nt+1 while computing nt
                const uint4* p = (const uint4*)pWh[nt + 1];
                bH[nxt].q[0] = p[0]; bH[nxt].q[1] = p[1];
            }
            pWh[nt] += 16;            // this nt's hi pointer consumed for this K-step
            #pragma unroll
            for (int m = 0; m < MT; ++m)
                acc[m][nt] = __builtin_amdgcn_wmma_f32_16x16x32_bf16(
                    false, aH[m].b, false, bH[cur].b, (short)0, acc[m][nt], false, false);
            #pragma unroll
            for (int m = 0; m < MT; ++m)
                acc[m][nt] = __builtin_amdgcn_wmma_f32_16x16x32_bf16(
                    false, aL[m].b, false, bH[cur].b, (short)0, acc[m][nt], false, false);
            #pragma unroll
            for (int m = 0; m < MT; ++m)
                acc[m][nt] = __builtin_amdgcn_wmma_f32_16x16x32_bf16(
                    false, aH[m].b, false, bL.b, (short)0, acc[m][nt], false, false);
        }
        // double buffer: next iteration stages the other LDS tile, so no
        // trailing barrier is needed here.
    }

    // ---- epilogue 1: deterministic column partial sums (bias added later) --
    // D layout: lane holds col = 80w + 16nt + mrow, rows {16m + 8*half + v}.
    // In-lane sum covers 64 of 128 rows; xor-16 shuffle completes the column.
    #pragma unroll
    for (int nt = 0; nt < NT_WAVE; ++nt) {
        float s = 0.0f;
        #pragma unroll
        for (int m = 0; m < MT; ++m)
            #pragma unroll
            for (int v = 0; v < 8; ++v)
                s += acc[m][nt][v];
        s += __shfl_xor(s, 16, 32);
        if (lane < 16)
            partial[(size_t)blockIdx.x * NCOL + wave * 80 + nt * 16 + lane] = s;
    }

    // ---- epilogue 2: per-wave argmax(acc + b + gumbel) over its 80 columns -
    float bv[NT_WAVE];
    #pragma unroll
    for (int nt = 0; nt < NT_WAVE; ++nt)
        bv[nt] = bvec[wave * 80 + nt * 16 + mrow];

    #pragma unroll
    for (int m = 0; m < MT; ++m) {
        #pragma unroll
        for (int v = 0; v < 8; ++v) {
            const int    row = m * 16 + half * 8 + v;
            const float* g   = gumbel + (size_t)(r0 + row) * NCOL;
            float best = -1e30f;
            int   bidx = 0;
            #pragma unroll
            for (int nt = 0; nt < NT_WAVE; ++nt) {
                const int   c   = wave * 80 + nt * 16 + mrow;
                const float val = acc[m][nt][v] + bv[nt] + g[c];
                if (val > best || (val == best && c < bidx)) { best = val; bidx = c; }
            }
            // reduce across the 16 lanes of this half-group (same row)
            #pragma unroll
            for (int off = 1; off < 16; off <<= 1) {
                float ov = __shfl_xor(best, off, 32);
                int   oi = __shfl_xor(bidx, off, 32);
                if (ov > best || (ov == best && oi < bidx)) { best = ov; bidx = oi; }
            }
            if (mrow == 0) { cval[row][wave] = best; cidx[row][wave] = bidx; }
        }
    }
    __syncthreads();

    // ---- epilogue 3: cross-wave argmax reduce + vars gather (16 rows/wave) -
    for (int j = 0; j < BM / 8; ++j) {
        const int row = wave * (BM / 8) + j;
        float best = (lane < 8) ? cval[row][lane] : -1e30f;
        int   bidx = (lane < 8) ? cidx[row][lane] : 0x7fffffff;
        #pragma unroll
        for (int off = 1; off < 8; off <<= 1) {
            float ov = __shfl_xor(best, off, 32);
            int   oi = __shfl_xor(bidx, off, 32);
            if (ov > best || (ov == best && oi < bidx)) { best = ov; bidx = oi; }
        }
        bidx = __shfl(bidx, 0, 32);

        const int kg = bidx / ENTRYV;   // codebook group of the one-hot
        const int ev = bidx % ENTRYV;
        const size_t  grow = (size_t)(r0 + row);
        float*        orow = out + grow * HID;
        const float4* vr   = (const float4*)(vars + (size_t)ev * VARD);
        float4*       dsel = (float4*)(orow + kg * VARD);
        float4*       dzer = (float4*)(orow + (1 - kg) * VARD);
        float4 z; z.x = z.y = z.z = z.w = 0.0f;
        for (int d = lane; d < VARD / 4; d += 32) {
            dsel[d] = vr[d];
            dzer[d] = z;
        }
    }
}

// ---------------------------------------------------------------------------
// Kernel 3: reduce per-block partials -> x2 mean (+bias) -> softmax -> y2
// ---------------------------------------------------------------------------
__global__ __launch_bounds__(NCOL)
void finalize_kernel(const float* __restrict__ partial,
                     const float* __restrict__ bvec,
                     float* __restrict__ y2) {
    __shared__ float red[NCOL];
    const int t = threadIdx.x;

    float s = 0.0f;
    for (int i = 0; i < NBLOCKS; ++i) s += partial[(size_t)i * NCOL + t];
    const float x2 = s * (1.0f / (float)BT_ROWS) + bvec[t];

    red[t] = x2;
    __syncthreads();
    for (int off = 512; off > 0; off >>= 1) {
        if (t < off && t + off < NCOL) red[t] = fmaxf(red[t], red[t + off]);
        __syncthreads();
    }
    const float mx = red[0];
    __syncthreads();

    const float e = __expf(x2 - mx);
    red[t] = e;
    __syncthreads();
    for (int off = 512; off > 0; off >>= 1) {
        if (t < off && t + off < NCOL) red[t] += red[t + off];
        __syncthreads();
    }
    y2[t] = e / red[0];
}

// ---------------------------------------------------------------------------
extern "C" void kernel_launch(void* const* d_in, const int* in_sizes, int n_in,
                              void* d_out, int out_size, void* d_ws, size_t ws_size,
                              hipStream_t stream) {
    const float* x      = (const float*)d_in[0];
    const float* W      = (const float*)d_in[1];
    const float* b      = (const float*)d_in[2];
    const float* vars   = (const float*)d_in[3];
    // d_in[4] = tau: unused — argmax is invariant to positive temperature and the
    // straight-through forward value is exactly the one-hot; softmax(tau) cancels.
    const float* gumbel = (const float*)d_in[5];

    float* out = (float*)d_out;
    float* y2  = out + (size_t)BT_ROWS * HID;

    // workspace layout: Wh | Wl | column partials  (~2.6 MB total)
    unsigned* Wh      = (unsigned*)d_ws;
    unsigned* Wl      = Wh + (size_t)NCOL * (HID / 2);
    float*    partial = (float*)(Wl + (size_t)NCOL * (HID / 2));

    wsplit_kernel<<<(NCOL * (HID / 2) + 255) / 256, 256, 0, stream>>>(W, Wh, Wl);
    qgemm_kernel<<<NBLOCKS, 256, 0, stream>>>(x, Wh, Wl, b, vars, gumbel, out, partial);
    finalize_kernel<<<1, NCOL, 0, stream>>>(partial, b, y2);
}
